// CDMoE_19344532702115
// MI455X (gfx1250) — matrix operand order repaired and checked by reading, not compiled, not run on hardware
//
#include <hip/hip_runtime.h>
#include <hip/hip_bf16.h>
#include <cstdint>

// ---------------------------------------------------------------------------
// CDMoE (product-key MoE) for MI455X / gfx1250, wave32 + WMMA bf16 +
// async global->LDS staging (ASYNCcnt path).
// Pipeline:
//   0) fp32 -> bf16 casts (X, W_up, W_down, W_q)
//   1) H1 = silu(X @ W_up^T + b_up)          WMMA bf16, out bf16
//   2) H  = H1 @ W_down^T + b_down           WMMA bf16, out f32 + bf16
//   3) Q  = H @ W_q^T                        WMMA bf16, out f32
//   4) per-token routing (sim, 2x top-8, cartesian top-8, softmax),
//      z = silu(softmax * <h, down_embed>), out = sum z * up_embed
// ---------------------------------------------------------------------------

typedef __attribute__((ext_vector_type(16))) __bf16 bf16x16;
typedef __attribute__((ext_vector_type(8)))  __bf16 bf16x8;
typedef __attribute__((ext_vector_type(4)))  __bf16 bf16x4;
typedef __attribute__((ext_vector_type(8)))  float  v8f;
typedef __attribute__((ext_vector_type(4)))  float  f32x4;

union FragAB { bf16x16 v; bf16x8 h[2]; };

#define TOKENS   4096
#define D_MODEL  2048
#define D_CROSS  8192
#define D_PRIV   1024
#define D_QTOT   1024
#define N_HEADS  8
#define NUM_KEYS 128
#define KPH      8

// Async copy 16B global -> LDS (GLOBAL_LOAD_ASYNC_TO_LDS_B128, ASYNCcnt).
// LDS address VGPR carries the wave-relative LDS byte offset == low 32 bits
// of the generic pointer to a __shared__ object (ISA 10.2 aperture rules).
__device__ __forceinline__ void async_cp16(const __bf16* g, __bf16* l) {
  uint32_t lds = (uint32_t)(uintptr_t)l;
  uint64_t ga  = (uint64_t)(uintptr_t)g;
  asm volatile("global_load_async_to_lds_b128 %0, %1, off"
               :: "v"(lds), "v"(ga) : "memory");
}
__device__ __forceinline__ void wait_async_le6() {
  asm volatile("s_wait_asynccnt 6" ::: "memory");
}
__device__ __forceinline__ void wait_async_0() {
  asm volatile("s_wait_asynccnt 0" ::: "memory");
}

// ---------------------------------------------------------------------------
// fp32 -> bf16 cast, 4 elements / thread (n must be a multiple of 1024)
// ---------------------------------------------------------------------------
__global__ __launch_bounds__(256) void cvt_f32_to_bf16(
    const float* __restrict__ s, __bf16* __restrict__ d, long n) {
  long i = ((long)blockIdx.x * 256 + threadIdx.x) * 4;
  if (i + 4 <= n) {
    f32x4 v = *(const f32x4*)(s + i);
    bf16x4 o;
    o[0] = (__bf16)v[0]; o[1] = (__bf16)v[1];
    o[2] = (__bf16)v[2]; o[3] = (__bf16)v[3];
    *(bf16x4*)(d + i) = o;
  }
}

// ---------------------------------------------------------------------------
// WMMA bf16 GEMM:  C[M,N] = A[M,K] @ B[N,K]^T (+bias) (+silu)
// Block = 256 thr (8 waves as 2x4), macro-tile 128x256, wave tile 64x64,
// BK=32, 16 WMMAs per wave per K-step (1 ds_load_b128 per WMMA).
// Double-buffered LDS filled by async global->LDS; fragments per ISA 7.12.2:
//   A-frag: lane(0..15)=row, hi-half K split {0-7|16-23} / {8-15|24-31}
//   B-frag: lane(0..15)=col, hi-half K split {0-7|16-23} / {8-15|24-31}
// ---------------------------------------------------------------------------
template<bool BIAS, bool SILU, bool OUT_BF, bool OUT_F32>
__global__ __launch_bounds__(256) void gemm_wmma_bf16(
    const __bf16* __restrict__ A, const __bf16* __restrict__ B,
    const float* __restrict__ bias,
    __bf16* __restrict__ Cb, float* __restrict__ Cf,
    int M, int N, int K) {
  __shared__ __bf16 sA[2][128][48];   // 48 = 32 + pad, keeps 16B alignment
  __shared__ __bf16 sB[2][256][48];

  const int tid  = threadIdx.x;
  const int wave = tid >> 5;
  const int lane = tid & 31;
  const int wr   = wave >> 2;      // 0..1
  const int wc   = wave & 3;       // 0..3
  const int r    = lane & 15;
  const int hi   = lane >> 4;

  const int m0 = blockIdx.y * 128;
  const int n0 = blockIdx.x * 256;

  v8f acc[4][4];
  #pragma unroll
  for (int mi = 0; mi < 4; ++mi)
    #pragma unroll
    for (int ni = 0; ni < 4; ++ni)
      acc[mi][ni] = (v8f){0.f,0.f,0.f,0.f,0.f,0.f,0.f,0.f};

  const int lrow = tid >> 1;            // 0..127  (A staging row)
  const int lcol = (tid & 1) << 4;      // 0 or 16 (A staging col)

  // issue one K-tile (2 A + 4 B async 16B copies / thread) into buffer `buf`
  auto issue_tile = [&](int k0, int buf) {
    const __bf16* gA = A + (size_t)(m0 + lrow) * K + k0 + lcol;
    const __bf16* gB = B + (size_t)(n0 + tid) * K + k0;
    async_cp16(gA,      &sA[buf][lrow][lcol]);
    async_cp16(gA + 8,  &sA[buf][lrow][lcol + 8]);
    async_cp16(gB,      &sB[buf][tid][0]);
    async_cp16(gB + 8,  &sB[buf][tid][8]);
    async_cp16(gB + 16, &sB[buf][tid][16]);
    async_cp16(gB + 24, &sB[buf][tid][24]);
    if (k0 + 64 < K) {                  // warm L2 for tile after next
      __builtin_prefetch((const void*)(gA + 64), 0, 0);
      __builtin_prefetch((const void*)(gB + 64), 0, 0);
    }
  };

  // compute one K-step from buffer `buf`
  auto compute_tile = [&](int buf) {
    FragAB af[4];
    #pragma unroll
    for (int mi = 0; mi < 4; ++mi) {
      const int rr = wr * 64 + mi * 16 + r;
      const int kb = hi * 8;
      af[mi].h[0] = *(const bf16x8*)&sA[buf][rr][kb];
      af[mi].h[1] = *(const bf16x8*)&sA[buf][rr][kb + 16];
    }
    #pragma unroll
    for (int ni = 0; ni < 4; ++ni) {
      const int cc = wc * 64 + ni * 16 + r;
      const int kb = hi * 8;
      FragAB bf;
      bf.h[0] = *(const bf16x8*)&sB[buf][cc][kb];
      bf.h[1] = *(const bf16x8*)&sB[buf][cc][kb + 16];
      #pragma unroll
      for (int mi = 0; mi < 4; ++mi)
        acc[mi][ni] = __builtin_amdgcn_wmma_f32_16x16x32_bf16(
            false, af[mi].v, false, bf.v, (short)0, acc[mi][ni],
            false, false);
    }
  };

  const int nk = K >> 5;                // >= 32 for all our shapes
  issue_tile(0, 0);

  for (int i = 0; i < nk - 1; ++i) {    // straight-line pipelined body
    const int buf = i & 1;
    issue_tile((i + 1) << 5, buf ^ 1);  // fetch next tile into other buffer
    wait_async_le6();                   // tile i landed (in-order completion)
    __syncthreads();                    // tile i visible to all waves
    compute_tile(buf);
    __syncthreads();                    // done reading buf before re-fill
  }
  wait_async_0();
  __syncthreads();
  compute_tile((nk - 1) & 1);

  // ---- epilogue: C layout VGPR v -> M = v + 8*hi, N = lane&15 (ISA 7.12.2)
  #pragma unroll
  for (int mi = 0; mi < 4; ++mi) {
    #pragma unroll
    for (int ni = 0; ni < 4; ++ni) {
      const int col = n0 + wc * 64 + ni * 16 + r;
      float bia = 0.f;
      if (BIAS) bia = bias[col];
      #pragma unroll
      for (int v = 0; v < 8; ++v) {
        const int row = m0 + wr * 64 + mi * 16 + v + 8 * hi;
        float val = acc[mi][ni][v] + bia;
        if (SILU) val = val / (1.f + __expf(-val));
        if (OUT_BF) Cb[(size_t)row * N + col] = (__bf16)val;
        if (OUT_F32) Cf[(size_t)row * N + col] = val;
      }
    }
  }
}

// ---------------------------------------------------------------------------
// Routing + expert gather. One block per token, wave h handles head h.
// ---------------------------------------------------------------------------
__global__ __launch_bounds__(256) void moe_route_kernel(
    const float* __restrict__ H,          // [T, D_PRIV]
    const float* __restrict__ Q,          // [T, D_QTOT]
    const float* __restrict__ keys,       // [8,128,2,64]
    const float* __restrict__ down_embed, // [16384, D_PRIV]
    const float* __restrict__ up_embed,   // [16384, D_MODEL]
    float* __restrict__ out) {            // [T, D_MODEL]
  const int t    = blockIdx.x;
  const int tid  = threadIdx.x;
  const int h    = tid >> 5;
  const int lane = tid & 31;

  __shared__ float s_out[D_MODEL];
  __shared__ float s_sim[N_HEADS][2][NUM_KEYS];
  __shared__ float s_sx[N_HEADS][KPH], s_sy[N_HEADS][KPH];
  __shared__ int   s_ix[N_HEADS][KPH], s_iy[N_HEADS][KPH];
  __shared__ float s_sel_s[N_HEADS][KPH];
  __shared__ int   s_sel_i[N_HEADS][KPH];

  for (int i = tid; i < D_MODEL; i += 256) s_out[i] = 0.f;

  // ---- sim[p][h][k] = <q[p,h,:], keys[h,k,p,:]> ; 4 keys per lane
  const float* qtok = Q + (size_t)t * D_QTOT;
  for (int p = 0; p < 2; ++p) {
    const float* qp = qtok + p * 512 + h * 64;
    for (int kk = 0; kk < 4; ++kk) {
      const int k = lane * 4 + kk;
      const float* kv = keys + (((size_t)h * NUM_KEYS + k) * 2 + p) * 64;
      float acc = 0.f;
      #pragma unroll 8
      for (int n = 0; n < 64; ++n) acc += qp[n] * kv[n];
      s_sim[h][p][k] = acc;
    }
  }
  __syncthreads();

  // ---- top-8 per half via iterative wave argmax (ties -> lowest index)
  for (int p = 0; p < 2; ++p) {
    float v[4]; int id[4];
    #pragma unroll
    for (int j = 0; j < 4; ++j) { id[j] = lane + 32 * j; v[j] = s_sim[h][p][id[j]]; }
    for (int sel = 0; sel < KPH; ++sel) {
      float bm = v[0]; int bi = id[0];
      #pragma unroll
      for (int j = 1; j < 4; ++j)
        if (v[j] > bm || (v[j] == bm && id[j] < bi)) { bm = v[j]; bi = id[j]; }
      float m = bm; int mi = bi;
      #pragma unroll
      for (int off = 16; off > 0; off >>= 1) {
        float om = __shfl_xor(m, off, 32);
        int   oi = __shfl_xor(mi, off, 32);
        if (om > m || (om == m && oi < mi)) { m = om; mi = oi; }
      }
      if (lane == 0) {
        if (p) { s_sy[h][sel] = m; s_iy[h][sel] = mi; }
        else   { s_sx[h][sel] = m; s_ix[h][sel] = mi; }
      }
      if (lane == (mi & 31)) v[mi >> 5] = -1e30f;
    }
  }
  __syncthreads();

  // ---- cartesian 8x8 -> top-8 combined
  float cv[2];
  #pragma unroll
  for (int j = 0; j < 2; ++j) {
    const int pr = lane * 2 + j;
    cv[j] = s_sx[h][pr >> 3] + s_sy[h][pr & 7];
  }
  for (int sel = 0; sel < KPH; ++sel) {
    float bm = cv[0]; int bp = lane * 2;
    if (cv[1] > bm) { bm = cv[1]; bp = lane * 2 + 1; }
    float m = bm; int mp = bp;
    #pragma unroll
    for (int off = 16; off > 0; off >>= 1) {
      float om = __shfl_xor(m, off, 32);
      int   op = __shfl_xor(mp, off, 32);
      if (om > m || (om == m && op < mp)) { m = om; mp = op; }
    }
    if (lane == 0) {
      s_sel_s[h][sel] = m;
      s_sel_i[h][sel] = s_ix[h][mp >> 3] * NUM_KEYS + s_iy[h][mp & 7];
    }
    if ((mp >> 1) == lane) cv[mp & 1] = -1e30f;
  }
  __syncthreads();

  // ---- softmax over selected scores (descending order -> [0] is max)
  float w[KPH]; float ssum = 0.f;
  const float smax = s_sel_s[h][0];
  #pragma unroll
  for (int j = 0; j < KPH; ++j) { w[j] = __expf(s_sel_s[h][j] - smax); ssum += w[j]; }
  const float inv = 1.f / ssum;

  // ---- z_k = silu(softmax_k * <h, down_embed[e]>), out += z_k * up_embed[e]
  const float* hv = H + (size_t)t * D_PRIV;
  for (int sel = 0; sel < KPH; ++sel) {
    const int e = s_sel_i[h][sel];
    const float* de = down_embed + (size_t)e * D_PRIV;
    float acc = 0.f;
    #pragma unroll 4
    for (int d = lane; d < D_PRIV; d += 32) acc += hv[d] * de[d];
    #pragma unroll
    for (int off = 16; off > 0; off >>= 1) acc += __shfl_xor(acc, off, 32);
    float zk = acc * (w[sel] * inv);
    zk = zk / (1.f + __expf(-zk));
    const float* ue = up_embed + (size_t)e * D_MODEL;
    for (int d = lane; d < D_MODEL; d += 32)
      atomicAdd(&s_out[d], zk * ue[d]);   // ds_add_f32
  }
  __syncthreads();

  for (int i = tid; i < D_MODEL; i += 256)
    out[(size_t)t * D_MODEL + i] = s_out[i];
}

// ---------------------------------------------------------------------------
extern "C" void kernel_launch(void* const* d_in, const int* in_sizes, int n_in,
                              void* d_out, int out_size, void* d_ws, size_t ws_size,
                              hipStream_t stream) {
  const float* X      = (const float*)d_in[0];  // [2,2048,2048]
  const float* W_up   = (const float*)d_in[1];  // [8192,2048]
  const float* b_up   = (const float*)d_in[2];
  const float* W_down = (const float*)d_in[3];  // [1024,8192]
  const float* b_down = (const float*)d_in[4];
  const float* W_q    = (const float*)d_in[5];  // [1024,1024]
  const float* keys   = (const float*)d_in[6];  // [8,128,2,64]
  const float* dn_emb = (const float*)d_in[7];  // [16384,1024]
  const float* up_emb = (const float*)d_in[8];  // [16384,2048]
  float* out = (float*)d_out;

  char* ws = (char*)d_ws;
  size_t off = 0;
  auto alloc = [&](size_t bytes) -> void* {
    void* p = ws + off; off += (bytes + 255) & ~(size_t)255; return p;
  };
  __bf16* Xb   = (__bf16*)alloc((size_t)TOKENS * D_MODEL * 2);
  __bf16* Wupb = (__bf16*)alloc((size_t)D_CROSS * D_MODEL * 2);
  __bf16* Wdnb = (__bf16*)alloc((size_t)D_PRIV * D_CROSS * 2);
  __bf16* Wqb  = (__bf16*)alloc((size_t)D_QTOT * D_PRIV * 2);
  __bf16* H1b  = (__bf16*)alloc((size_t)TOKENS * D_CROSS * 2);
  float*  Hf   = (float*) alloc((size_t)TOKENS * D_PRIV * 4);
  __bf16* Hb   = (__bf16*)alloc((size_t)TOKENS * D_PRIV * 2);
  float*  Qf   = (float*) alloc((size_t)TOKENS * D_QTOT * 4);

  // 0) fp32 -> bf16 casts
  {
    long nX = (long)TOKENS * D_MODEL;
    long nU = (long)D_CROSS * D_MODEL;
    long nD = (long)D_PRIV * D_CROSS;
    long nQ = (long)D_QTOT * D_PRIV;
    cvt_f32_to_bf16<<<(int)(nX / 1024), 256, 0, stream>>>(X, Xb, nX);
    cvt_f32_to_bf16<<<(int)(nU / 1024), 256, 0, stream>>>(W_up, Wupb, nU);
    cvt_f32_to_bf16<<<(int)(nD / 1024), 256, 0, stream>>>(W_down, Wdnb, nD);
    cvt_f32_to_bf16<<<(int)(nQ / 1024), 256, 0, stream>>>(W_q, Wqb, nQ);
  }

  // 1) H1 = silu(X @ W_up^T + b_up)   [4096, 8192] bf16
  {
    dim3 g(D_CROSS / 256, TOKENS / 128);
    gemm_wmma_bf16<true, true, true, false><<<g, 256, 0, stream>>>(
        Xb, Wupb, b_up, H1b, nullptr, TOKENS, D_CROSS, D_MODEL);
  }
  // 2) H = H1 @ W_down^T + b_down     [4096, 1024] f32 + bf16
  {
    dim3 g(D_PRIV / 256, TOKENS / 128);
    gemm_wmma_bf16<true, false, true, true><<<g, 256, 0, stream>>>(
        H1b, Wdnb, b_down, Hb, Hf, TOKENS, D_PRIV, D_CROSS);
  }
  // 3) Q = H @ W_q^T                  [4096, 1024] f32
  {
    dim3 g(D_QTOT / 256, TOKENS / 128);
    gemm_wmma_bf16<false, false, false, true><<<g, 256, 0, stream>>>(
        Hb, Wqb, nullptr, nullptr, Qf, TOKENS, D_QTOT, D_PRIV);
  }
  // 4) routing + gather -> out
  moe_route_kernel<<<TOKENS, 256, 0, stream>>>(Hf, Qf, keys, dn_emb, up_emb, out);
}
